// GPTLanguageModel_62792421867529
// MI455X (gfx1250) — matrix operand order repaired and checked by reading, not compiled, not run on hardware
//
#include <hip/hip_runtime.h>
#include <hip/hip_bf16.h>

// ---------------- model constants (from reference) ----------------
#define L_    8
#define D_    384
#define H_    8
#define HS_   48
#define T_    128
#define B_    32
#define V_    32000
#define NTOK  (B_ * T_)     // 4096 token rows
#define DFF   (4 * D_)      // 1536
#define EPS_  1e-5f

typedef __attribute__((ext_vector_type(16))) _Float16 v16h;
typedef __attribute__((ext_vector_type(8)))  float    v8f;

// ---------------- CDNA5 async global->LDS copy (ASYNCcnt path) ----------------
__device__ __forceinline__ void async_copy_b128(unsigned lds_off, const void* gptr) {
  // copies 16 bytes global -> LDS, tracked by ASYNCcnt
  asm volatile("global_load_async_to_lds_b128 %0, %1, off"
               :: "v"(lds_off), "v"(gptr) : "memory");
}
__device__ __forceinline__ void wait_async0() {
  asm volatile("s_wait_asynccnt 0" ::: "memory");
}

// ---------------- f16-A (async) / f32-B (cvt at stage) WMMA GEMM ----------------
// C[M,N] = act( A16[M,K] @ B[K,N] + bias[N] ),  C (f32) and/or C16 (f16) outputs.
// Geometry: block tile 128x128, 8 waves, wave tile 32x64 => 8 WMMA / K-slab of 32.
// Requirements (all satisfied by this model): M%128==0, N%128==0, K%32==0.
#define BM 128
#define BN 128
#define BK 32
#define KPAD 8   // LDS row stride 40 halfs = 80B (16B-aligned rows)

__global__ __launch_bounds__(256) void gemm_wmma_f16(
    const _Float16* __restrict__ A, const float* __restrict__ Bm,
    const float* __restrict__ bias, float* __restrict__ C,
    _Float16* __restrict__ C16, int M, int N, int K, int relu) {
  __shared__ _Float16 As[2][BM][BK + KPAD];  // 2 x 10 KB, row-major [m][k]
  __shared__ _Float16 Bs[2][BN][BK + KPAD];  // 2 x 10 KB, transposed [n][k]

  const int tid   = threadIdx.x;
  const int lane  = tid & 31;
  const int wave  = tid >> 5;       // 0..7
  const int waveM = wave & 3;       // 4 waves along M (32 rows each)
  const int waveN = wave >> 2;      // 2 waves along N (64 cols each)
  const int m0 = blockIdx.x * BM;
  const int n0 = blockIdx.y * BN;

  // fragment-gather coordinates (16-bit A/B WMMA layouts, wave32)
  const int frow  = lane & 15;      // row within 16 (A: M, B: N)
  const int khalf = lane >> 4;      // 0 / 1
  const int koffA = khalf * 8;      // A: elems0-7 -> K=koffA.., elems8-15 -> K=16+koffA..
  const int koffB = khalf * 16;     // unused; B is gathered with koffA, in the same K order as A

  v8f acc[2][4] = {};

  // staging coordinates
  const int arow = tid >> 1;               // 0..127
  const int aseg = (tid & 1) * 16;         // 0 / 16 (f16 elements)
  const int bkr  = tid >> 3;               // 0..31
  const int bnc  = (tid & 7) * 16;         // 0..112

  // ---- stage one K-slab into LDS buffer `buf` (A async f16, B cvt f32->f16)
  auto stage = [&](int buf, int k0) {
    const _Float16* ga = A + (size_t)(m0 + arow) * K + (k0 + aseg);
    const unsigned  la = (unsigned)(unsigned long long)&As[buf][arow][aseg];
    async_copy_b128(la,      ga);                              // 16B
    async_copy_b128(la + 16, (const char*)ga + 16);            // 16B
    const float* gb = Bm + (size_t)(k0 + bkr) * N + (n0 + bnc);
#pragma unroll
    for (int j = 0; j < 16; ++j) Bs[buf][bnc + j][bkr] = (_Float16)gb[j];
  };

  // ---- 8x v_wmma_f32_16x16x32_f16 on LDS buffer `buf`
  auto compute = [&](int buf) {
    v16h af[2], bf[4];
#pragma unroll
    for (int mi = 0; mi < 2; ++mi) {
      const int r = waveM * 32 + mi * 16 + frow;
#pragma unroll
      for (int j = 0; j < 8; ++j) {
        af[mi][j]     = As[buf][r][koffA + j];
        af[mi][8 + j] = As[buf][r][16 + koffA + j];
      }
    }
#pragma unroll
    for (int ni = 0; ni < 4; ++ni) {
      const int c = waveN * 64 + ni * 16 + frow;
#pragma unroll
      for (int j = 0; j < 16; ++j) bf[ni][j] = Bs[buf][c][(j & 8) * 2 + koffA + (j & 7)];
    }
#pragma unroll
    for (int mi = 0; mi < 2; ++mi)
#pragma unroll
      for (int ni = 0; ni < 4; ++ni)
        acc[mi][ni] = __builtin_amdgcn_wmma_f32_16x16x32_f16(
            false, af[mi], false, bf[ni], (short)0, acc[mi][ni], false, false);
  };

  // ---- software pipeline: peel the tail so the hot loop has no branches
  stage(0, 0);
  int buf = 0;
  for (int k0 = 0; k0 < K - BK; k0 += BK) {
    wait_async0();        // my async copies into `buf` have landed
    __syncthreads();      // everyone's copies + B ds-stores have landed
    stage(buf ^ 1, k0 + BK);   // overlap next slab with compute
    compute(buf);
    buf ^= 1;
  }
  wait_async0();
  __syncthreads();
  compute(buf);

  // ---- epilogue: C layout => lane<16: N=lane, M=r ; lane>=16: N=lane-16, M=8+r
  const int crow = khalf * 8;
  const int ccol = frow;
#pragma unroll
  for (int mi = 0; mi < 2; ++mi)
#pragma unroll
    for (int ni = 0; ni < 4; ++ni) {
      const int gcol = n0 + waveN * 64 + ni * 16 + ccol;
      const float bb = bias ? bias[gcol] : 0.0f;
#pragma unroll
      for (int r = 0; r < 8; ++r) {
        const int grow = m0 + waveM * 32 + mi * 16 + crow + r;
        float v = acc[mi][ni][r] + bb;
        if (relu) v = v > 0.0f ? v : 0.0f;
        if (C)   C[(size_t)grow * N + gcol]   = v;
        if (C16) C16[(size_t)grow * N + gcol] = (_Float16)v;
      }
    }
}

// ---------------- embedding gather (writes f32 stream + f16 shadow) ----------------
__global__ __launch_bounds__(128) void embed_kernel(
    const int* __restrict__ idx, const float* __restrict__ tok,
    const float* __restrict__ pos, float* __restrict__ x,
    _Float16* __restrict__ x16) {
  const int row = blockIdx.x;          // 0..NTOK-1
  const int t   = row % T_;
  const int v   = idx[row];
#pragma unroll
  for (int i = 0; i < 3; ++i) {
    const int c = threadIdx.x + i * 128;
    const float val = tok[(size_t)v * D_ + c] + pos[(size_t)t * D_ + c];
    x[(size_t)row * D_ + c]   = val;
    x16[(size_t)row * D_ + c] = (_Float16)val;
  }
}

// ---------------- causal attention, online softmax (1 thread = 1 query row) ----
// Writes f16 directly (its only consumer is the Wo GEMM's A operand).
__global__ __launch_bounds__(128) void attention_kernel(
    const float* __restrict__ q, const float* __restrict__ k,
    const float* __restrict__ v, _Float16* __restrict__ o) {
  __shared__ float ks[T_][HS_ + 1];
  __shared__ float vs[T_][HS_ + 1];
  const int bh = blockIdx.x;
  const int b  = bh / H_;
  const int h  = bh % H_;
  const int t  = threadIdx.x;           // query row

  const size_t base = ((size_t)(b * T_ + t)) * D_ + h * HS_;
#pragma unroll
  for (int d = 0; d < HS_; ++d) {
    ks[t][d] = k[base + d];
    vs[t][d] = v[base + d];
  }
  __syncthreads();

  const float scale = 0.14433756729740645f;   // 48^-0.5
  float qr[HS_];
#pragma unroll
  for (int d = 0; d < HS_; ++d) qr[d] = q[base + d] * scale;

  float m = -INFINITY, sum = 0.0f;
  float acc[HS_];
#pragma unroll
  for (int d = 0; d < HS_; ++d) acc[d] = 0.0f;

  for (int s = 0; s <= t; ++s) {
    float x = 0.0f;
#pragma unroll
    for (int d = 0; d < HS_; ++d) x += qr[d] * ks[s][d];
    const float nm   = fmaxf(m, x);
    const float corr = __expf(m - nm);
    const float p    = __expf(x - nm);
    sum = sum * corr + p;
#pragma unroll
    for (int d = 0; d < HS_; ++d) acc[d] = acc[d] * corr + p * vs[s][d];
    m = nm;
  }
  const float inv = 1.0f / sum;
#pragma unroll
  for (int d = 0; d < HS_; ++d) o[base + d] = (_Float16)(acc[d] * inv);
}

// ------------- fused residual add + LayerNorm (in place on x) + f16 shadow -------------
__global__ __launch_bounds__(128) void add_ln_kernel(
    float* __restrict__ x, const float* __restrict__ y,
    const float* __restrict__ g, const float* __restrict__ bta,
    _Float16* __restrict__ x16) {
  __shared__ float red[128];
  const int row = blockIdx.x;
  const int tid = threadIdx.x;
  float* xr = x + (size_t)row * D_;
  const float* yr = y ? y + (size_t)row * D_ : nullptr;

  float vals[3];
  float s = 0.0f;
#pragma unroll
  for (int i = 0; i < 3; ++i) {
    const int c = tid + i * 128;
    float t = xr[c] + (yr ? yr[c] : 0.0f);
    vals[i] = t;
    s += t;
  }
  red[tid] = s;
  __syncthreads();
  for (int off = 64; off > 0; off >>= 1) {
    if (tid < off) red[tid] += red[tid + off];
    __syncthreads();
  }
  const float mean = red[0] * (1.0f / D_);
  __syncthreads();

  float vsum = 0.0f;
#pragma unroll
  for (int i = 0; i < 3; ++i) {
    const float c = vals[i] - mean;
    vsum += c * c;
  }
  red[tid] = vsum;
  __syncthreads();
  for (int off = 64; off > 0; off >>= 1) {
    if (tid < off) red[tid] += red[tid + off];
    __syncthreads();
  }
  const float rstd = rsqrtf(red[0] * (1.0f / D_) + EPS_);
#pragma unroll
  for (int i = 0; i < 3; ++i) {
    const int c = tid + i * 128;
    const float o = (vals[i] - mean) * rstd * g[c] + bta[c];
    xr[c] = o;
    x16[(size_t)row * D_ + c] = (_Float16)o;
  }
}

// ---------------- launcher ----------------
extern "C" void kernel_launch(void* const* d_in, const int* in_sizes, int n_in,
                              void* d_out, int out_size, void* d_ws, size_t ws_size,
                              hipStream_t stream) {
  (void)in_sizes; (void)n_in; (void)out_size; (void)ws_size;

  const int*   idx  = (const int*)  d_in[0];
  const float* tok  = (const float*)d_in[1];
  const float* pos  = (const float*)d_in[2];
  const float* Wq   = (const float*)d_in[3];
  const float* Wk   = (const float*)d_in[4];
  const float* Wv   = (const float*)d_in[5];
  const float* Wo   = (const float*)d_in[6];
  const float* bo   = (const float*)d_in[7];
  const float* W1   = (const float*)d_in[8];
  const float* b1   = (const float*)d_in[9];
  const float* W2   = (const float*)d_in[10];
  const float* b2   = (const float*)d_in[11];
  const float* ln1g = (const float*)d_in[12];
  const float* ln1b = (const float*)d_in[13];
  const float* ln2g = (const float*)d_in[14];
  const float* ln2b = (const float*)d_in[15];
  const float* lnfg = (const float*)d_in[16];
  const float* lnfb = (const float*)d_in[17];
  const float* lmW  = (const float*)d_in[18];
  const float* lmb  = (const float*)d_in[19];
  float* out = (float*)d_out;

  // workspace partition
  const size_t ND = (size_t)NTOK * D_;
  float* ws = (float*)d_ws;
  float* x  = ws;            // f32 residual stream
  float* y  = x  + ND;       // f32 sublayer output
  float* q  = y  + ND;
  float* kk = q  + ND;
  float* vv = kk + ND;
  _Float16* x16  = (_Float16*)(vv + ND);     // f16 shadow of x (GEMM A operand)
  _Float16* at16 = x16 + ND;                 // f16 attention output
  _Float16* h16  = at16 + ND;                // f16 FFN hidden, NTOK*DFF

  const dim3 blk256(256), blk128(128);
  const dim3 gD(NTOK / BM, D_ / BN);    // (32, 3)
  const dim3 gF(NTOK / BM, DFF / BN);   // (32, 12)
  const dim3 gV(NTOK / BM, V_ / BN);    // (32, 250)

  embed_kernel<<<NTOK, blk128, 0, stream>>>(idx, tok, pos, x, x16);

  for (int l = 0; l < L_; ++l) {
    const float* wq = Wq + (size_t)l * D_ * D_;
    const float* wk = Wk + (size_t)l * D_ * D_;
    const float* wv = Wv + (size_t)l * D_ * D_;
    const float* wo = Wo + (size_t)l * D_ * D_;
    const float* w1 = W1 + (size_t)l * D_ * DFF;
    const float* w2 = W2 + (size_t)l * DFF * D_;

    gemm_wmma_f16<<<gD, blk256, 0, stream>>>(x16, wq, nullptr, q,  nullptr, NTOK, D_, D_, 0);
    gemm_wmma_f16<<<gD, blk256, 0, stream>>>(x16, wk, nullptr, kk, nullptr, NTOK, D_, D_, 0);
    gemm_wmma_f16<<<gD, blk256, 0, stream>>>(x16, wv, nullptr, vv, nullptr, NTOK, D_, D_, 0);
    attention_kernel<<<B_ * H_, blk128, 0, stream>>>(q, kk, vv, at16);
    gemm_wmma_f16<<<gD, blk256, 0, stream>>>(at16, wo, bo + (size_t)l * D_, y, nullptr,
                                             NTOK, D_, D_, 0);
    add_ln_kernel<<<NTOK, blk128, 0, stream>>>(x, y, ln1g + (size_t)l * D_,
                                               ln1b + (size_t)l * D_, x16);
    gemm_wmma_f16<<<gF, blk256, 0, stream>>>(x16, w1, b1 + (size_t)l * DFF, nullptr, h16,
                                             NTOK, DFF, D_, 1);
    gemm_wmma_f16<<<gD, blk256, 0, stream>>>(h16, w2, b2 + (size_t)l * D_, y, nullptr,
                                             NTOK, D_, DFF, 0);
    add_ln_kernel<<<NTOK, blk128, 0, stream>>>(x, y, ln2g + (size_t)l * D_,
                                               ln2b + (size_t)l * D_, x16);
  }

  add_ln_kernel<<<NTOK, blk128, 0, stream>>>(x, nullptr, lnfg, lnfb, x16);

  // LM head: the dominant GEMM (4096 x 32000 x 384) straight into d_out
  gemm_wmma_f16<<<gV, blk256, 0, stream>>>(x16, lmW, lmb, out, nullptr, NTOK, V_, D_, 0);
}
